// DyGANGenerator_87170656239863
// MI455X (gfx1250) — matrix operand order repaired and checked by reading, not compiled, not run on hardware
//
#include <hip/hip_runtime.h>
#include <hip/hip_bf16.h>

// ---------------------------------------------------------------------------
// DyGAN generator on gfx1250: 16-step LSTM decode.
// All GEMMs via v_wmma_f32_16x16x32_f16 with fragment-packed B operands,
// padded-K A operands (no tail guards -> straight-line b128 loads).
// ---------------------------------------------------------------------------

typedef __attribute__((ext_vector_type(16))) _Float16 v16h;
typedef __attribute__((ext_vector_type(8)))  _Float16 v8h;
typedef __attribute__((ext_vector_type(8)))  float    v8f;

constexpr int BATCH = 256;
constexpr int ZDIM  = 128;
constexpr int ZP1   = 129;
constexpr int ZPAD  = 160;            // ZP1 padded to multiple of 32
constexpr int HIN   = 128;
constexpr int H     = 512;
constexpr int H4    = 4 * H;          // 2048
constexpr int HT    = 64;
constexpr int NNODE = 10000;
constexpr int NPAD  = 10016;          // NNODE padded to multiple of 32
constexpr int RW    = 16;
constexpr int XHLD  = HIN + H;        // 640: [x | h] concat, row-major

enum { ACT_NONE = 0, ACT_RELU = 1, ACT_TANH = 2 };

__device__ __forceinline__ float sigmoidf_(float x) {
    return 1.0f / (1.0f + __expf(-x));
}

// ---------------------------------------------------------------------------
// WMMA GEMM: C[M, N] = act(A[M, Kpad] @ B[Kpad, N] + bias)
//  - A: row-major f16, lda >= Kpad (Kpad multiple of 32, zero-padded cols)
//  - Bp: fragment-packed f16: [(tn * Kc + kc) * 32 + lane] * 16 halves,
//        lane half j holds B(k = kc*32 + j + (lane>=16 ? 16:0), n = tn*16 + lane%16)
//  - one wave computes TN adjacent 16x16 C tiles (shares the A fragment)
// ---------------------------------------------------------------------------
template <int ACT, int TN>
__global__ __launch_bounds__(256) void gemm_wmma_f16(
    const _Float16* __restrict__ A, int lda,
    const _Float16* __restrict__ Bp,
    const float* __restrict__ bias,
    float* __restrict__ Cf, int ldcf,
    _Float16* __restrict__ Ch, int ldch,
    int M, int N, int Kpad)
{
    const int wid  = (blockIdx.x * blockDim.x + threadIdx.x) >> 5;
    const int lane = threadIdx.x & 31;
    const int tiles_n  = N >> 4;
    const int groups_n = tiles_n / TN;
    const int ngroups  = (M >> 4) * groups_n;
    if (wid >= ngroups) return;                  // whole-wave uniform exit

    const int tm  = wid / groups_n;
    const int tn0 = (wid % groups_n) * TN;
    const int r16 = lane & 15;
    const bool hi = lane >= 16;
    const int Kc  = Kpad >> 5;

    // A: one contiguous 32B span (16 K) per 32-K chunk per lane
    const _Float16* __restrict__ Arow =
        A + (size_t)(tm * 16 + r16) * lda + (hi ? 16 : 0);
    // B: 32B contiguous fragment per lane per (tile, chunk)
    const _Float16* __restrict__ Bl = Bp + (size_t)lane * 16;

    v8f acc[TN];
#pragma unroll
    for (int t = 0; t < TN; ++t) acc[t] = (v8f){};

    for (int kc = 0; kc < Kc; ++kc) {
        // prefetch chunk kc+2 (two ahead) of first B tile (speculative, dropped if OOB)
        __builtin_prefetch(Bl + (((size_t)tn0 * Kc + kc + 2) << 9), 0, 1);

        const v8h a0 = *(const v8h*)(Arow + (kc << 5));
        const v8h a1 = *(const v8h*)(Arow + (kc << 5) + 8);
        v16h af;
#pragma unroll
        for (int j = 0; j < 8; ++j) { af[j] = a0[j]; af[8 + j] = a1[j]; }

#pragma unroll
        for (int t = 0; t < TN; ++t) {
            const v16h bf =
                *(const v16h*)(Bl + (((size_t)(tn0 + t) * Kc + kc) << 9));
            acc[t] = __builtin_amdgcn_wmma_f32_16x16x32_f16(
                /*neg_a=*/false, af, /*neg_b=*/false, bf,
                /*c_mod=*/(short)0, acc[t], /*reuse_a=*/false, /*reuse_b=*/false);
        }
    }

    // Epilogue. C/D layout: lane holds N=r16; VGPR v holds M=v (+8 hi lanes)
#pragma unroll
    for (int t = 0; t < TN; ++t) {
        const int n = (tn0 + t) * 16 + r16;
        const float bv = bias ? bias[n] : 0.f;
#pragma unroll
        for (int v = 0; v < 8; ++v) {
            const int m = tm * 16 + v + (hi ? 8 : 0);
            float val = acc[t][v] + bv;
            if (ACT == ACT_RELU) val = fmaxf(val, 0.f);
            if (ACT == ACT_TANH) val = tanhf(val);
            if (Cf) Cf[(size_t)m * ldcf + n] = val;
            if (Ch) Ch[(size_t)m * ldch + n] = (_Float16)val;
        }
    }
}

// ---------------------------------------------------------------------------
// Pack a row-major fp32 weight [K, N] (optionally a concat of src0[0:K1] and
// src1[K1:K]) into the WMMA B fragment layout with K zero-padded to Kpad.
__global__ void pack_b_kernel(const float* __restrict__ src0,
                              const float* __restrict__ src1, int K1,
                              _Float16* __restrict__ dst,
                              int K, int N, int Kpad)
{
    const int i = blockIdx.x * blockDim.x + threadIdx.x;
    const int total = N * Kpad;          // == tiles_n * Kc * 32 * 16
    if (i >= total) return;
    const int j    = i & 15;             // half index within lane
    const int lane = (i >> 4) & 31;
    const int cidx = i >> 9;             // tn * Kc + kc
    const int Kc   = Kpad >> 5;
    const int tn   = cidx / Kc;
    const int kc   = cidx % Kc;
    const int n = tn * 16 + (lane & 15);
    const int k = kc * 32 + j + (lane >= 16 ? 16 : 0);
    float v = 0.f;
    if (k < K) {
        v = (k < K1 || !src1) ? src0[(size_t)k * N + n]
                              : src1[(size_t)(k - K1) * N + n];
    }
    dst[i] = (_Float16)v;
}

// noise [256,129] fp32 -> f16 [256,160] zero-padded
__global__ void pad_noise_kernel(const float* __restrict__ src,
                                 _Float16* __restrict__ dst)
{
    const int i = blockIdx.x * blockDim.x + threadIdx.x;
    if (i >= BATCH * ZPAD) return;
    const int b = i / ZPAD, col = i % ZPAD;
    dst[i] = (col < ZP1) ? (_Float16)src[(size_t)b * ZP1 + col] : (_Float16)0.f;
}

// xh[:, :HIN] = f16(x0);  last_t = 0
__global__ void init_state_kernel(const float* __restrict__ x0,
                                  _Float16* __restrict__ xh,
                                  float* __restrict__ last_t)
{
    const int i = blockIdx.x * blockDim.x + threadIdx.x;
    if (i < BATCH * HIN) {
        const int b = i / HIN, j = i % HIN;
        xh[(size_t)b * XHLD + j] = (_Float16)x0[i];
    }
    if (i < BATCH) last_t[i] = 0.f;
}

// LSTM pointwise gate update; writes h fp32 and h f16 into xh[:, HIN:]
__global__ void lstm_update_kernel(const float* __restrict__ gates,
                                   float* __restrict__ c,
                                   float* __restrict__ h,
                                   _Float16* __restrict__ xh)
{
    const int i = blockIdx.x * blockDim.x + threadIdx.x;
    if (i >= BATCH * H) return;
    const int b = i / H, j = i % H;
    const float* g = gates + (size_t)b * H4;
    const float ig = sigmoidf_(g[j]);
    const float fg = sigmoidf_(g[H + j]);
    const float gg = tanhf(g[2 * H + j]);
    const float og = sigmoidf_(g[3 * H + j]);
    const float cv = fg * c[i] + ig * gg;
    const float hv = og * tanhf(cv);
    c[i] = cv;
    h[i] = hv;
    xh[(size_t)b * XHLD + HIN + j] = (_Float16)hv;
}

// Row softmax with temperature 0.5 (logits*2); writes p to rw[b, s, :] (fp32)
// and to p16 [256, NPAD] (f16, zero-padded) for the p @ Wup GEMM.
__global__ __launch_bounds__(256) void softmax_row_kernel(
    const float* __restrict__ logits,
    float* __restrict__ rw,
    _Float16* __restrict__ p16, int s)
{
    __shared__ float red[256];
    const int b = blockIdx.x, t = threadIdx.x;
    const float* row = logits + (size_t)b * NNODE;

    float mx = -3.4e38f;
    for (int n = t; n < NNODE; n += 256) mx = fmaxf(mx, row[n]);
    red[t] = mx; __syncthreads();
    for (int off = 128; off > 0; off >>= 1) {
        if (t < off) red[t] = fmaxf(red[t], red[t + off]);
        __syncthreads();
    }
    mx = red[0]; __syncthreads();

    float sum = 0.f;
    for (int n = t; n < NNODE; n += 256) sum += __expf(2.f * (row[n] - mx));
    red[t] = sum; __syncthreads();
    for (int off = 128; off > 0; off >>= 1) {
        if (t < off) red[t] += red[t + off];
        __syncthreads();
    }
    const float inv = 1.f / red[0];

    float* out = rw + (size_t)b * RW * NNODE + (size_t)s * NNODE;
    _Float16* pr = p16 + (size_t)b * NPAD;
    for (int n = t; n < NNODE; n += 256) {
        const float p = __expf(2.f * (row[n] - mx)) * inv;
        out[n] = p;
        pr[n]  = (_Float16)p;
    }
    if (t < NPAD - NNODE) pr[NNODE + t] = (_Float16)0.f;   // zero K-pad
}

// t[b] = clamp(sum_j td[b,j]*Wtp[j], last_t[b], 1.0); writes ts[b, s]
__global__ void time_step_kernel(const float* __restrict__ td,
                                 const float* __restrict__ Wtp,
                                 float* __restrict__ ts_out,
                                 float* __restrict__ last_t, int s)
{
    const int b = blockIdx.x * blockDim.x + threadIdx.x;
    if (b >= BATCH) return;
    const float* r = td + (size_t)b * HT;
    float acc = 0.f;
#pragma unroll
    for (int j = 0; j < HT; ++j) acc += r[j] * Wtp[j];
    const float lt = last_t[b];
    const float t = fminf(1.f, fmaxf(acc, lt));
    ts_out[(size_t)b * RW + s] = t;
    last_t[b] = t;
}

__global__ void copy_f32_kernel(const float* __restrict__ src,
                                float* __restrict__ dst, int n)
{
    const int i = blockIdx.x * blockDim.x + threadIdx.x;
    if (i < n) dst[i] = src[i];
}

// ---------------------------------------------------------------------------
static void launch_gemm(int act, int tn,
                        const _Float16* A, int lda, const _Float16* Bp,
                        const float* bias, float* Cf, int ldcf,
                        _Float16* Ch, int ldch,
                        int M, int N, int Kpad, hipStream_t stream)
{
    const int ngroups = (M / 16) * ((N / 16) / tn);
    dim3 grid((ngroups + 7) / 8), block(256);
    if (tn == 5) {
        gemm_wmma_f16<ACT_NONE, 5><<<grid, block, 0, stream>>>(
            A, lda, Bp, bias, Cf, ldcf, Ch, ldch, M, N, Kpad);
        return;
    }
    switch (act) {
    case ACT_NONE:
        gemm_wmma_f16<ACT_NONE, 4><<<grid, block, 0, stream>>>(
            A, lda, Bp, bias, Cf, ldcf, Ch, ldch, M, N, Kpad);
        break;
    case ACT_RELU:
        gemm_wmma_f16<ACT_RELU, 4><<<grid, block, 0, stream>>>(
            A, lda, Bp, bias, Cf, ldcf, Ch, ldch, M, N, Kpad);
        break;
    default:
        gemm_wmma_f16<ACT_TANH, 4><<<grid, block, 0, stream>>>(
            A, lda, Bp, bias, Cf, ldcf, Ch, ldch, M, N, Kpad);
        break;
    }
}

static inline void pack_b(const float* s0, const float* s1, int K1,
                          _Float16* dst, int K, int N, int Kpad, hipStream_t s)
{
    const int total = N * Kpad;
    pack_b_kernel<<<(total + 255) / 256, 256, 0, s>>>(s0, s1, K1, dst, K, N, Kpad);
}

extern "C" void kernel_launch(void* const* d_in, const int* in_sizes, int n_in,
                              void* d_out, int out_size, void* d_ws, size_t ws_size,
                              hipStream_t stream)
{
    // ---- inputs (setup_inputs dict order) ----
    const float* noise   = (const float*)d_in[0];
    const float* hidden0 = (const float*)d_in[1];
    const float* W_comp  = (const float*)d_in[2];
    const float* b_comp  = (const float*)d_in[3];
    const float* W_sh    = (const float*)d_in[4];
    const float* b_sh    = (const float*)d_in[5];
    const float* W_ci    = (const float*)d_in[6];
    const float* b_ci    = (const float*)d_in[7];
    const float* W_hi    = (const float*)d_in[8];
    const float* b_hi    = (const float*)d_in[9];
    const float* W_ih    = (const float*)d_in[10];
    const float* W_hh    = (const float*)d_in[11];
    const float* b_lstm  = (const float*)d_in[12];
    const float* Wd1     = (const float*)d_in[13];
    const float* bd1     = (const float*)d_in[14];
    const float* Wd2     = (const float*)d_in[15];
    const float* bd2     = (const float*)d_in[16];
    const float* Wup     = (const float*)d_in[17];
    const float* bup     = (const float*)d_in[18];
    const float* Wtd     = (const float*)d_in[19];
    const float* btd     = (const float*)d_in[20];
    const float* Wtp     = (const float*)d_in[21];

    // ---- outputs: rw [B,RW,N] fp32, ts [B,RW,1] fp32, h [B,H] fp32 ----
    float* rw_out = (float*)d_out;
    float* ts_out = rw_out + (size_t)BATCH * RW * NNODE;
    float* h_out  = ts_out + (size_t)BATCH * RW;

    // ---- workspace carve-up ----
    char* ws = (char*)d_ws;
    size_t off = 0;
    auto alloc = [&](size_t bytes) -> void* {
        void* p = ws + off;
        off += (bytes + 255) & ~(size_t)255;
        return p;
    };
    // fragment-packed f16 weights (size = N * Kpad halves each)
    _Float16* wihhP  = (_Float16*)alloc(sizeof(_Float16) * H4 * XHLD);     // [640,2048]
    _Float16* wcompP = (_Float16*)alloc(sizeof(_Float16) * ZDIM * ZPAD);   // [129->160,128]
    _Float16* wshP   = (_Float16*)alloc(sizeof(_Float16) * H * ZDIM);      // [128,512]
    _Float16* wciP   = (_Float16*)alloc(sizeof(_Float16) * H * H);
    _Float16* whiP   = (_Float16*)alloc(sizeof(_Float16) * H * H);
    _Float16* wd1P   = (_Float16*)alloc(sizeof(_Float16) * H * H);
    _Float16* wd2P   = (_Float16*)alloc(sizeof(_Float16) * NNODE * H);     // [512,10000]
    _Float16* wupP   = (_Float16*)alloc(sizeof(_Float16) * HIN * NPAD);    // [10000->10016,128]
    _Float16* wtdP   = (_Float16*)alloc(sizeof(_Float16) * HT * H);        // [512,64]
    // f16 activations
    _Float16* noise16 = (_Float16*)alloc(sizeof(_Float16) * BATCH * ZPAD);
    _Float16* z16     = (_Float16*)alloc(sizeof(_Float16) * BATCH * ZDIM);
    _Float16* inter16 = (_Float16*)alloc(sizeof(_Float16) * BATCH * H);
    _Float16* a16     = (_Float16*)alloc(sizeof(_Float16) * BATCH * H);
    _Float16* p16     = (_Float16*)alloc(sizeof(_Float16) * BATCH * NPAD);
    _Float16* xh16    = (_Float16*)alloc(sizeof(_Float16) * BATCH * XHLD); // [x|h]
    // f32 activations / state
    float* gates  = (float*)alloc(sizeof(float) * BATCH * H4);
    float* logits = (float*)alloc(sizeof(float) * BATCH * NNODE);
    float* cbuf   = (float*)alloc(sizeof(float) * BATCH * H);
    float* hbuf   = (float*)alloc(sizeof(float) * BATCH * H);
    float* tdbuf  = (float*)alloc(sizeof(float) * BATCH * HT);
    float* last_t = (float*)alloc(sizeof(float) * BATCH);
    (void)ws_size; (void)in_sizes; (void)n_in; (void)out_size;

    // ---- weight packing (every call: deterministic, mostly L2-resident) ----
    pack_b(W_ih, W_hh, HIN, wihhP,  XHLD,  H4,    XHLD, stream); // concat [640,2048]
    pack_b(W_comp, nullptr, 0, wcompP, ZP1, ZDIM, ZPAD, stream);
    pack_b(W_sh,   nullptr, 0, wshP,   ZDIM, H,   ZDIM, stream);
    pack_b(W_ci,   nullptr, 0, wciP,   H, H,      H,    stream);
    pack_b(W_hi,   nullptr, 0, whiP,   H, H,      H,    stream);
    pack_b(Wd1,    nullptr, 0, wd1P,   H, H,      H,    stream);
    pack_b(Wd2,    nullptr, 0, wd2P,   H, NNODE,  H,    stream);
    pack_b(Wup,    nullptr, 0, wupP,   NNODE, HIN, NPAD, stream);
    pack_b(Wtd,    nullptr, 0, wtdP,   H, HT,     H,    stream);
    pad_noise_kernel<<<(BATCH * ZPAD + 255) / 256, 256, 0, stream>>>(noise, noise16);

    // ---- initial state ----
    // z = noise @ W_comp + b_comp                       [256,128] Kpad=160
    launch_gemm(ACT_NONE, 4, noise16, ZPAD, wcompP, b_comp,
                nullptr, 0, z16, ZDIM, BATCH, ZDIM, ZPAD, stream);
    // inter = tanh(z @ W_sh + b_sh)                     [256,512]
    launch_gemm(ACT_TANH, 4, z16, ZDIM, wshP, b_sh,
                nullptr, 0, inter16, H, BATCH, H, ZDIM, stream);
    // c0 = tanh(inter @ W_ci + b_ci)                    fp32 state
    launch_gemm(ACT_TANH, 4, inter16, H, wciP, b_ci,
                cbuf, H, nullptr, 0, BATCH, H, H, stream);
    // h0 = tanh(inter @ W_hi + b_hi)  -> fp32 h and f16 into xh[:, HIN:]
    launch_gemm(ACT_TANH, 4, inter16, H, whiP, b_hi,
                hbuf, H, xh16 + HIN, XHLD, BATCH, H, H, stream);
    // x0 = hidden0 -> xh[:, :HIN]; last_t = 0
    init_state_kernel<<<(BATCH * HIN + 255) / 256, 256, 0, stream>>>(hidden0, xh16, last_t);

    // ---- 16 decode steps ----
    for (int s = 0; s < RW; ++s) {
        // gates = [x|h] @ [W_ih;W_hh] + b_lstm          [256,2048] K=640
        launch_gemm(ACT_NONE, 4, xh16, XHLD, wihhP, b_lstm,
                    gates, H4, nullptr, 0, BATCH, H4, XHLD, stream);
        // pointwise LSTM; updates c/h fp32, h f16 into xh[:, HIN:]
        lstm_update_kernel<<<(BATCH * H + 255) / 256, 256, 0, stream>>>(gates, cbuf, hbuf, xh16);
        // a = relu(h @ Wd1 + bd1) -> f16                [256,512]
        launch_gemm(ACT_RELU, 4, xh16 + HIN, XHLD, wd1P, bd1,
                    nullptr, 0, a16, H, BATCH, H, H, stream);
        // logits = a @ Wd2 + bd2                        [256,10000] K=512, TN=5
        launch_gemm(ACT_NONE, 5, a16, H, wd2P, bd2,
                    logits, NNODE, nullptr, 0, BATCH, NNODE, H, stream);
        // p = softmax(logits / 0.5) -> rw[:, s, :] fp32 and p16 (padded)
        softmax_row_kernel<<<BATCH, 256, 0, stream>>>(logits, rw_out, p16, s);
        // x = p @ Wup + bup -> f16 into xh[:, :HIN]     [256,128] Kpad=10016
        launch_gemm(ACT_NONE, 4, p16, NPAD, wupP, bup,
                    nullptr, 0, xh16, XHLD, BATCH, HIN, NPAD, stream);
        // td = tanh(h @ Wtd + btd)                      [256,64]
        launch_gemm(ACT_TANH, 4, xh16 + HIN, XHLD, wtdP, btd,
                    tdbuf, HT, nullptr, 0, BATCH, HT, H, stream);
        // t = clamp(td @ Wtp, last_t, 1) -> ts[:, s]
        time_step_kernel<<<(BATCH + 255) / 256, 256, 0, stream>>>(tdbuf, Wtp, ts_out, last_t, s);
    }

    // final hidden state -> output
    copy_f32_kernel<<<(BATCH * H + 255) / 256, 256, 0, stream>>>(hbuf, h_out, BATCH * H);
}
